// MAB_51866025066892
// MI455X (gfx1250) — matrix operand order repaired and checked by reading, not compiled, not run on hardware
//
#include <hip/hip_runtime.h>

#define DEVINL __device__ __forceinline__

typedef __attribute__((ext_vector_type(16))) __bf16 v16bf;
typedef __attribute__((ext_vector_type(8)))  float  v8f;

constexpr int kB  = 8;
constexpr int kN  = 1024;
constexpr int kD  = 256;
constexpr int kH  = 8;
constexpr int kDh = 32;

// ---------------- WMMA wrapper ----------------
DEVINL v8f wmma_bf16(v16bf a, v16bf b, v8f c) {
  return __builtin_amdgcn_wmma_f32_16x16x32_bf16(
      /*neg_a=*/false, a, /*neg_b=*/false, b,
      /*c_mod=*/(short)0, c, /*reuse_a=*/false, /*reuse_b=*/false);
}

// ---------------- CDNA5 async global->LDS copy (ASYNCcnt path) --------------
// LDS address is the vdst VGPR value plus the instruction offset; generic LDS ptr low 32 bits
// are the LDS byte offset, so truncation gives the vdst value.
DEVINL unsigned lds_off(const void* p) { return (unsigned)(uintptr_t)p; }

DEVINL void async_ld_b128(unsigned lds_byte_off, const void* gaddr) {
  asm volatile("global_load_async_to_lds_b128 %0, %1, off"
               :: "v"(lds_byte_off), "v"(gaddr) : "memory");
}
DEVINL void wait_async0() {
  asm volatile("s_wait_asynccnt 0" ::: "memory");
}

// ---------------- Operand loaders (contiguous k = 16*(lane/16) + e, same in A and B) ----
// A 16x32: lane&15 = row; K = e (+16 for upper lanes)
DEVINL v16bf ldA_f32(const float* p, int ld, int m0, int k0, int lane) {
  int row = m0 + (lane & 15);
  int kb  = k0 + ((lane >> 4) << 4);
  const float* s = p + (size_t)row * ld + kb;
  v16bf a;
#pragma unroll
  for (int e = 0; e < 16; ++e) {
    a[e] = (__bf16)s[e];
  }
  return a;
}

DEVINL v16bf ldA_bf(const __bf16* p, int ld, int m0, int k0, int lane) {
  int row = m0 + (lane & 15);
  int kb  = k0 + ((lane >> 4) << 4);
  const __bf16* s = p + (size_t)row * ld + kb;
  v16bf a;
#pragma unroll
  for (int e = 0; e < 16; ++e) {
    a[e] = s[e];
  }
  return a;
}

// B 32x16 (row-major B[k][n]): lane&15 = column; K = e (+16 for upper lanes)
DEVINL v16bf ldB_f32(const float* p, int ld, int k0, int n0, int lane) {
  int col = n0 + (lane & 15);
  int kb  = k0 + ((lane >> 4) << 4);
  const float* s = p + (size_t)kb * ld + col;
  v16bf b;
#pragma unroll
  for (int e = 0; e < 16; ++e) b[e] = (__bf16)s[(size_t)e * ld];
  return b;
}

DEVINL v16bf ldB_bf(const __bf16* p, int ld, int k0, int n0, int lane) {
  int col = n0 + (lane & 15);
  int kb  = k0 + ((lane >> 4) << 4);
  const __bf16* s = p + (size_t)kb * ld + col;
  v16bf b;
#pragma unroll
  for (int e = 0; e < 16; ++e) b[e] = s[(size_t)e * ld];
  return b;
}

// B operand taken from a row-major matrix used transposed: B[k][n] = p[n*ld+k]
// (contiguous 32B per lane -> vectorizable global_load)
DEVINL v16bf ldBT_bf(const __bf16* p, int ld, int n0, int k0, int lane) {
  int col = n0 + (lane & 15);
  int kb  = k0 + ((lane >> 4) << 4);
  const __bf16* s = p + (size_t)col * ld + kb;
  v16bf b;
#pragma unroll
  for (int e = 0; e < 16; ++e) b[e] = s[e];
  return b;
}

// ---------------- Kernel 1: projection GEMM (X@W + bias) * mask -------------
// Block (128 thr = 4 waves) computes a 64x64 tile; per K-step both operand
// tiles are staged in LDS with async b128 copies, each wave does 4 WMMAs.
__global__ __launch_bounds__(128) void proj_gemm_kernel(
    const float* __restrict__ X, const float* __restrict__ W,
    const float* __restrict__ bias, const int* __restrict__ lengths,
    __bf16* __restrict__ outb, float* __restrict__ outf) {
  __shared__ float As[64][36];   // 144B row stride (16B aligned), pad kills conflicts
  __shared__ float Bs[32][64];

  int t = threadIdx.x;
  int lane = t & 31;
  int w = t >> 5;
  int m0  = (blockIdx.x >> 2) << 6;
  int n0b = (blockIdx.x & 3) << 6;
  int b   = m0 >> 10;            // 64-row tiles never straddle a batch
  int len = lengths[b];

  unsigned aBase = lds_off(&As[0][0]);
  unsigned bBase = lds_off(&Bs[0][0]);

  v8f acc[4] = {{}, {}, {}, {}};

  for (int k0 = 0; k0 < kD; k0 += 32) {
    // A tile 64x32 f32: 512 16B chunks (row = c/8, seg = c%8)
#pragma unroll
    for (int i = 0; i < 4; ++i) {
      int c = t + (i << 7);
      int row = c >> 3, seg = c & 7;
      async_ld_b128(aBase + (unsigned)(row * 144 + seg * 16),
                    X + (size_t)(m0 + row) * kD + k0 + seg * 4);
    }
    // B tile 32x64 f32: 512 16B chunks (row = c/16, seg = c%16)
#pragma unroll
    for (int i = 0; i < 4; ++i) {
      int c = t + (i << 7);
      int row = c >> 4, seg = c & 15;
      async_ld_b128(bBase + (unsigned)(row * 256 + seg * 16),
                    W + (size_t)(k0 + row) * kD + n0b + seg * 4);
    }
    wait_async0();
    __syncthreads();

    v16bf a = ldA_f32(&As[0][0], 36, w << 4, 0, lane);
#pragma unroll
    for (int nt = 0; nt < 4; ++nt) {
      v16bf bb = ldB_f32(&Bs[0][0], 64, 0, nt << 4, lane);
      acc[nt] = wmma_bf16(a, bb, acc[nt]);
    }
    __syncthreads();
  }

  int rb = m0 + (w << 4) + ((lane >> 4) << 3);
#pragma unroll
  for (int nt = 0; nt < 4; ++nt) {
    int col = n0b + (nt << 4) + (lane & 15);
    float bv = bias[col];
#pragma unroll
    for (int r = 0; r < 8; ++r) {
      int row = rb + r;
      float m = ((row & (kN - 1)) < len) ? 1.0f : 0.0f;
      float v = (acc[nt][r] + bv) * m;
      size_t idx = (size_t)row * kD + col;
      outb[idx] = (__bf16)v;
      if (outf) outf[idx] = v;
    }
  }
}

// ---------------- Kernel 2: attention (per b,h, 16-query tile) --------------
__global__ __launch_bounds__(128) void attn_kernel(
    const __bf16* __restrict__ qbf, const __bf16* __restrict__ kbf,
    const __bf16* __restrict__ vbf, const float* __restrict__ qf,
    const int* __restrict__ lengths, float* __restrict__ O) {
  __shared__ float Pt[4][16][33];   // per-wave P transpose staging

  int lane = threadIdx.x & 31;
  int wv   = threadIdx.x >> 5;
  int wave = (blockIdx.x << 2) | wv;
  int qt = wave & 63;
  int h  = (wave >> 6) & 7;
  int b  = wave >> 9;               // all 4 waves of a block share b -> len
  int q0 = qt << 4;
  int len = lengths[b];

  size_t base = (size_t)b * kN * kD + (size_t)h * kDh;
  const __bf16* qp = qbf + base;
  const __bf16* kp = kbf + base;
  const __bf16* vp = vbf + base;

  v16bf aQ = ldA_bf(qp, kD, q0, 0, lane);   // 16x32 query tile, reused

  v8f acc0 = {}, acc1 = {};
  float rs[8];
#pragma unroll
  for (int r = 0; r < 8; ++r) rs[r] = 0.0f;

  int rloc = (lane >> 4) << 3;   // 0 or 8 (row group per C-layout)
  int cl   = lane & 15;
  float (*P)[33] = Pt[wv];

  for (int j0 = 0; j0 < len; j0 += 32) {
    __syncthreads();             // WAR vs previous iteration's P reads
#pragma unroll
    for (int t = 0; t < 2; ++t) {
      int kt = j0 + (t << 4);
      v16bf bK = ldBT_bf(kp, kD, kt, 0, lane);   // K tile, transposed use
      v8f cz = {};
      v8f s8 = wmma_bf16(aQ, bK, cz);            // one WMMA = full 16x16 score
      int key = kt + cl;
      bool km = key < len;
#pragma unroll
      for (int r = 0; r < 8; ++r) {
        int row = q0 + rloc + r;
        float pv = (km && (row < len)) ? __expf(s8[r] * 0.0625f) : 0.0f;
        rs[r] += pv;
        P[rloc + r][(t << 4) + cl] = pv;
      }
    }
    __syncthreads();             // RAW: cross-lane P visibility
    v16bf aP  = ldA_f32(&P[0][0], 33, 0, 0, lane);
    v16bf bV0 = ldB_bf(vp, kD, j0, 0, lane);
    acc0 = wmma_bf16(aP, bV0, acc0);
    v16bf bV1 = ldB_bf(vp, kD, j0, 16, lane);
    acc1 = wmma_bf16(aP, bV1, acc1);
  }

  // Row sums: reduce across the 16 lanes of each half (masks 1,2,4,8 stay
  // inside a half, so lanes 0-15 reduce rows 0-7, lanes 16-31 rows 8-15).
#pragma unroll
  for (int r = 0; r < 8; ++r) {
    float s = rs[r];
    s += __shfl_xor(s, 1);
    s += __shfl_xor(s, 2);
    s += __shfl_xor(s, 4);
    s += __shfl_xor(s, 8);
    rs[r] = s;
  }

  const float* qr = qf + base;
  float* Ob = O + base;
#pragma unroll
  for (int r = 0; r < 8; ++r) {
    int row = q0 + rloc + r;
    float inv = 1.0f / (rs[r] + 1e-15f);
    size_t o = (size_t)row * kD + cl;
    Ob[o]      = qr[o]      + acc0[r] * inv;
    Ob[o + 16] = qr[o + 16] + acc1[r] * inv;
  }
}

// ---------------- Kernel 3/5: row LayerNorm (one wave per 256-wide row) -----
__global__ __launch_bounds__(128) void ln_kernel(
    const float* __restrict__ X, const float* __restrict__ g,
    const float* __restrict__ bta, float* __restrict__ Yf,
    __bf16* __restrict__ Ybf) {
  int lane = threadIdx.x & 31;
  size_t row = ((size_t)blockIdx.x << 2) | (threadIdx.x >> 5);
  const float* x = X + row * kD;

  float v[8];
  float s = 0.0f;
#pragma unroll
  for (int i = 0; i < 8; ++i) { v[i] = x[lane * 8 + i]; s += v[i]; }
  s += __shfl_xor(s, 1);  s += __shfl_xor(s, 2);
  s += __shfl_xor(s, 4);  s += __shfl_xor(s, 8);  s += __shfl_xor(s, 16);
  float mu = s * (1.0f / kD);

  float vs = 0.0f;
#pragma unroll
  for (int i = 0; i < 8; ++i) { float d = v[i] - mu; vs += d * d; }
  vs += __shfl_xor(vs, 1);  vs += __shfl_xor(vs, 2);
  vs += __shfl_xor(vs, 4);  vs += __shfl_xor(vs, 8);  vs += __shfl_xor(vs, 16);
  float rstd = rsqrtf(vs * (1.0f / kD) + 1e-5f);

#pragma unroll
  for (int i = 0; i < 8; ++i) {
    int c = lane * 8 + i;
    float y = (v[i] - mu) * rstd * g[c] + bta[c];
    Yf[row * kD + c] = y;
    if (Ybf) Ybf[row * kD + c] = (__bf16)y;
  }
}

// ---------------- Kernel 4: output proj + relu + residual -------------------
__global__ __launch_bounds__(128) void ffn_gemm_kernel(
    const __bf16* __restrict__ Xb, const float* __restrict__ W,
    const float* __restrict__ bias, const float* __restrict__ res,
    float* __restrict__ T) {
  __shared__ __bf16 Asb[64][48];  // 96B row stride (16B aligned)
  __shared__ float  Bs[32][64];

  int t = threadIdx.x;
  int lane = t & 31;
  int w = t >> 5;
  int m0  = (blockIdx.x >> 2) << 6;
  int n0b = (blockIdx.x & 3) << 6;

  unsigned aBase = lds_off(&Asb[0][0]);
  unsigned bBase = lds_off(&Bs[0][0]);

  v8f acc[4] = {{}, {}, {}, {}};

  for (int k0 = 0; k0 < kD; k0 += 32) {
    // A tile 64x32 bf16: 256 16B chunks (row = c/4, seg = c%4)
#pragma unroll
    for (int i = 0; i < 2; ++i) {
      int c = t + (i << 7);
      int row = c >> 2, seg = c & 3;
      async_ld_b128(aBase + (unsigned)(row * 96 + seg * 16),
                    Xb + (size_t)(m0 + row) * kD + k0 + seg * 8);
    }
    // B tile 32x64 f32: 512 16B chunks
#pragma unroll
    for (int i = 0; i < 4; ++i) {
      int c = t + (i << 7);
      int row = c >> 4, seg = c & 15;
      async_ld_b128(bBase + (unsigned)(row * 256 + seg * 16),
                    W + (size_t)(k0 + row) * kD + n0b + seg * 4);
    }
    wait_async0();
    __syncthreads();

    v16bf a = ldA_bf(&Asb[0][0], 48, w << 4, 0, lane);
#pragma unroll
    for (int nt = 0; nt < 4; ++nt) {
      v16bf bb = ldB_f32(&Bs[0][0], 64, 0, nt << 4, lane);
      acc[nt] = wmma_bf16(a, bb, acc[nt]);
    }
    __syncthreads();
  }

  int rb = m0 + (w << 4) + ((lane >> 4) << 3);
#pragma unroll
  for (int nt = 0; nt < 4; ++nt) {
    int col = n0b + (nt << 4) + (lane & 15);
    float bv = bias[col];
#pragma unroll
    for (int r = 0; r < 8; ++r) {
      size_t idx = (size_t)(rb + r) * kD + col;
      T[idx] = res[idx] + fmaxf(acc[nt][r] + bv, 0.0f);
    }
  }
}

// ---------------- Host launcher ---------------------------------------------
extern "C" void kernel_launch(void* const* d_in, const int* in_sizes, int n_in,
                              void* d_out, int out_size, void* d_ws, size_t ws_size,
                              hipStream_t stream) {
  const float* Q  = (const float*)d_in[0];
  const float* K  = (const float*)d_in[1];
  const int*   ln = (const int*)d_in[2];
  const float* Wq = (const float*)d_in[3];
  const float* bq = (const float*)d_in[4];
  const float* Wk = (const float*)d_in[5];
  const float* bk = (const float*)d_in[6];
  const float* Wv = (const float*)d_in[7];
  const float* bv = (const float*)d_in[8];
  const float* Wo = (const float*)d_in[9];
  const float* bo = (const float*)d_in[10];
  const float* g0 = (const float*)d_in[11];
  const float* b0 = (const float*)d_in[12];
  const float* g1 = (const float*)d_in[13];
  const float* b1 = (const float*)d_in[14];
  float* out = (float*)d_out;

  char* ws = (char*)d_ws;
  const size_t MB = 1u << 20;
  __bf16* qbf  = (__bf16*)(ws + 0);        // 4 MB
  __bf16* kbf  = (__bf16*)(ws + 4 * MB);   // 4 MB
  __bf16* vbf  = (__bf16*)(ws + 8 * MB);   // 4 MB
  float*  qf   = (float*)(ws + 12 * MB);   // 8 MB
  float*  Ob   = (float*)(ws + 20 * MB);   // 8 MB
  float*  ln0f = (float*)(ws + 0);         // reuse qbf+kbf (8 MB)
  __bf16* ln0b = (__bf16*)(ws + 8 * MB);   // reuse vbf (4 MB)
  float*  T    = (float*)(ws + 12 * MB);   // reuse qf (8 MB)

  dim3 blk(128);
  proj_gemm_kernel<<<512, blk, 0, stream>>>(Q, Wq, bq, ln, qbf, qf);
  proj_gemm_kernel<<<512, blk, 0, stream>>>(K, Wk, bk, ln, kbf, nullptr);
  proj_gemm_kernel<<<512, blk, 0, stream>>>(K, Wv, bv, ln, vbf, nullptr);
  attn_kernel<<<1024, blk, 0, stream>>>(qbf, kbf, vbf, qf, ln, Ob);
  ln_kernel<<<2048, blk, 0, stream>>>(Ob, g0, b0, ln0f, ln0b);
  ffn_gemm_kernel<<<512, blk, 0, stream>>>(ln0b, Wo, bo, ln0f, T);
  ln_kernel<<<2048, blk, 0, stream>>>(T, g1, b1, out, nullptr);
}
